// SparseAutoencoder_34385508172381
// MI455X (gfx1250) — compile-verified, matrix operand order checked by reading, not run on hardware
//
#include <hip/hip_runtime.h>
#include <hip/hip_bf16.h>
#include <math.h>

#define N_DIRS      32768
#define D_MODEL     768
#define BATCH       2048
#define TOPK        32
#define AUXK        128
#define DEAD_THRESH 256
#define AUXK_COEF   (1.0f / 32.0f)

typedef __attribute__((ext_vector_type(16))) __bf16 v16bf;
typedef __attribute__((ext_vector_type(8)))  __bf16 v8bf;
typedef __attribute__((ext_vector_type(4)))  __bf16 v4bf;
typedef __attribute__((ext_vector_type(8)))  float  v8f;

// ---------------------------------------------------------------------------
// Zero-init scratch accumulators (hits, musum, scalar sums)
// ---------------------------------------------------------------------------
__global__ void sae_zero(float* __restrict__ p, int n) {
    int i = blockIdx.x * 256 + threadIdx.x;
    if (i < n) p[i] = 0.0f;
}

// ---------------------------------------------------------------------------
// Bulk f32 -> bf16 conversion (enc_w), 4 elems/thread
// ---------------------------------------------------------------------------
__global__ __launch_bounds__(256) void sae_cvt_bf16(const float* __restrict__ src,
                                                    __bf16* __restrict__ dst,
                                                    long long n) {
    long long i = ((long long)blockIdx.x * 256 + threadIdx.x) * 4;
    if (i < n) {
        float4 v = *(const float4*)(src + i);
        v4bf o;
        o[0] = (__bf16)v.x; o[1] = (__bf16)v.y;
        o[2] = (__bf16)v.z; o[3] = (__bf16)v.w;
        *(v4bf*)(dst + i) = o;
    }
}

// ---------------------------------------------------------------------------
// xc_bf16 = bf16(x - pre_bias)   [BATCH, D_MODEL]
// ---------------------------------------------------------------------------
__global__ __launch_bounds__(256) void sae_make_xc(const float* __restrict__ x,
                                                   const float* __restrict__ pre_bias,
                                                   __bf16* __restrict__ xc_bf) {
    const int b = blockIdx.x, tid = threadIdx.x;
#pragma unroll
    for (int j = 0; j < 3; ++j) {
        int d = tid + j * 256;
        xc_bf[(size_t)b * D_MODEL + d] = (__bf16)(x[(size_t)b * D_MODEL + d] - pre_bias[d]);
    }
}

// ---------------------------------------------------------------------------
// Transpose dec_w [D_MODEL, N_DIRS] -> decT [N_DIRS, D_MODEL]
// ---------------------------------------------------------------------------
__global__ __launch_bounds__(256) void sae_transpose(const float* __restrict__ dec_w,
                                                     float* __restrict__ decT) {
    __shared__ float tile[32][33];
    const int tx = threadIdx.x & 31;
    const int ty = threadIdx.x >> 5;
    const int n0 = blockIdx.x * 32;
    const int d0 = blockIdx.y * 32;
#pragma unroll
    for (int j = 0; j < 4; ++j) {
        int d = d0 + ty + j * 8;
        tile[ty + j * 8][tx] = dec_w[(size_t)d * N_DIRS + n0 + tx];
    }
    __syncthreads();
#pragma unroll
    for (int j = 0; j < 4; ++j) {
        int n = n0 + ty + j * 8;
        decT[(size_t)n * D_MODEL + d0 + tx] = tile[tx][ty + j * 8];
    }
}

// ---------------------------------------------------------------------------
// latents = xc_bf16 @ enc_bf16^T + latent_bias  via v_wmma_f32_16x16x32_bf16
// Block: 256 threads = 8 waves, tile 16 rows x 512 cols.
// A tile (16x768 bf16, 24 KB) staged in LDS via async-to-LDS copies, shared by
// all 8 waves; each wave computes 16x64 with 4 accumulators.
// ---------------------------------------------------------------------------
__global__ __launch_bounds__(256) void sae_gemm_latents(
    const __bf16* __restrict__ xc_bf, const __bf16* __restrict__ enc_bf,
    const float* __restrict__ latent_bias, float* __restrict__ latents)
{
    __shared__ __bf16 lds_x[16 * D_MODEL];   // 24 KB

    const int tid    = threadIdx.x;
    const int lane   = tid & 31;
    const int mbase  = blockIdx.y * 16;
    const int nwbase = blockIdx.x * 512 + (tid >> 5) * 64;
    const int koff   = (lane & 16) ? 8 : 0;  // lanes 16-31 hold the +8 K-half
    const int l15    = lane & 15;

    // ---- async stage of the A tile into LDS (1536 x 16B chunks) ----
    {
        const unsigned lds_base =
            (unsigned)(unsigned long long)(void*)&lds_x[0];
        const unsigned long long gbase =
            (unsigned long long)(const void*)(xc_bf + (size_t)mbase * D_MODEL);
#pragma unroll
        for (int c = 0; c < 6; ++c) {
            int chunk = tid + c * 256;
            unsigned lds_off = lds_base + (unsigned)chunk * 16u;
            unsigned long long ga = gbase + (unsigned long long)chunk * 16ull;
            asm volatile("global_load_async_to_lds_b128 %0, %1, off"
                         :: "v"(lds_off), "v"(ga) : "memory");
        }
        asm volatile("s_wait_asynccnt 0x0" ::: "memory");
    }
    __syncthreads();

    v8f acc[4] = {};

#pragma unroll 2
    for (int kb = 0; kb < D_MODEL; kb += 32) {
        // A fragment from LDS: elems 0-7 = K kb+koff.., elems 8-15 = K kb+16+koff..
        const __bf16* arow = &lds_x[l15 * D_MODEL + kb + koff];
        v8bf alo = *(const v8bf*)arow;
        v8bf ahi = *(const v8bf*)(arow + 16);
        v16bf afrag = __builtin_shufflevector(alo, ahi,
            0, 1, 2, 3, 4, 5, 6, 7, 8, 9, 10, 11, 12, 13, 14, 15);

#pragma unroll
        for (int ns = 0; ns < 4; ++ns) {
            // B = enc^T: column n of B == row n of enc_bf (mirrors A layout)
            const __bf16* brow =
                enc_bf + (size_t)(nwbase + ns * 16 + l15) * D_MODEL + kb + koff;
            v8bf blo = *(const v8bf*)brow;
            v8bf bhi = *(const v8bf*)(brow + 16);
            v16bf bfrag = __builtin_shufflevector(blo, bhi,
                0, 1, 2, 3, 4, 5, 6, 7, 8, 9, 10, 11, 12, 13, 14, 15);

            acc[ns] = __builtin_amdgcn_wmma_f32_16x16x32_bf16(
                false, afrag, false, bfrag, (short)0, acc[ns], false, false);
        }
    }

    // C/D layout: lane l, reg r -> M = r + 8*(l>=16), N = l&15
    const int mrow = (lane & 16) ? 8 : 0;
#pragma unroll
    for (int ns = 0; ns < 4; ++ns) {
        int   ncol = nwbase + ns * 16 + l15;
        float bias = latent_bias[ncol];
        size_t base = (size_t)(mbase + mrow) * N_DIRS + ncol;
#pragma unroll
        for (int r = 0; r < 8; ++r)
            latents[base + (size_t)r * N_DIRS] = acc[ns][r] + bias;
    }
}

// ---------------------------------------------------------------------------
// Per-row top-k via row cached in LDS + iterative block argmax.
// mask == nullptr: main pass (also counts l0 positives and marks hits).
// mask != nullptr: auxk pass over latents * dead_mask.
// ---------------------------------------------------------------------------
__global__ __launch_bounds__(256) void sae_row_topk(
    const float* __restrict__ latents, const float* __restrict__ mask,
    float* __restrict__ out_vals, int* __restrict__ out_inds, int kcount,
    int* __restrict__ hits, float* __restrict__ l0sum)
{
    extern __shared__ float smem[];
    float* srow = smem;                    // N_DIRS floats (128 KB)
    float* rv   = smem + N_DIRS;           // 256 floats
    int*   ri   = (int*)(rv + 256);        // 256 ints

    const int b   = blockIdx.x;
    const int tid = threadIdx.x;
    const float* row = latents + (size_t)b * N_DIRS;

    int cnt = 0;
    if (mask) {
        for (int i = tid; i < N_DIRS; i += 256) srow[i] = row[i] * mask[i];
    } else {
        for (int i = tid; i < N_DIRS; i += 256) {
            float v = row[i];
            srow[i] = v;
            cnt += (v > 0.0f) ? 1 : 0;
        }
    }
    __syncthreads();

    if (!mask) {
        rv[tid] = (float)cnt;
        __syncthreads();
        for (int s = 128; s > 0; s >>= 1) {
            if (tid < s) rv[tid] += rv[tid + s];
            __syncthreads();
        }
        if (tid == 0) atomicAdd(l0sum, rv[0]);
        __syncthreads();
    }

    for (int it = 0; it < kcount; ++it) {
        float bm = -3.4e38f;
        int   bi = 0;
        for (int i = tid; i < N_DIRS; i += 256) {
            float v = srow[i];
            if (v > bm) { bm = v; bi = i; }
        }
        rv[tid] = bm; ri[tid] = bi;
        __syncthreads();
        for (int s = 128; s > 0; s >>= 1) {
            if (tid < s) {
                float ov = rv[tid + s]; int oi = ri[tid + s];
                if (ov > rv[tid] || (ov == rv[tid] && oi < ri[tid])) {
                    rv[tid] = ov; ri[tid] = oi;
                }
            }
            __syncthreads();
        }
        if (tid == 0) {
            out_vals[(size_t)b * kcount + it] = rv[0];
            out_inds[(size_t)b * kcount + it] = ri[0];
            if (hits && rv[0] > 0.001f) atomicAdd(&hits[ri[0]], 1);
            srow[ri[0]] = -3.4e38f;
        }
        __syncthreads();
    }
}

// ---------------------------------------------------------------------------
// new_stats = stats*(1 - clip(hits,0,1)) + 1 ; dead = new_stats > DEAD_THRESH
// ---------------------------------------------------------------------------
__global__ void sae_stats(const int* __restrict__ stats_in, const int* __restrict__ hits,
                          float* __restrict__ out_stats, float* __restrict__ dead) {
    int i = blockIdx.x * 256 + threadIdx.x;
    if (i < N_DIRS) {
        int fired = hits[i] > 0;
        int ns = fired ? 1 : (stats_in[i] + 1);
        out_stats[i] = (float)ns;
        dead[i] = (ns > DEAD_THRESH) ? 1.0f : 0.0f;
    }
}

// ---------------------------------------------------------------------------
// recons = sum_k relu(vals)*decT[inds] + pre_bias ; mse partials ; target ; mu sums
// ---------------------------------------------------------------------------
__global__ __launch_bounds__(256) void sae_decode_main(
    const float* __restrict__ x, const float* __restrict__ pre_bias,
    const float* __restrict__ decT,
    const float* __restrict__ vals, const int* __restrict__ inds,
    float* __restrict__ recons_out, float* __restrict__ target,
    float* __restrict__ musum, float* __restrict__ scalars /* [1]=msesum */)
{
    __shared__ float sval[TOPK];
    __shared__ int   sind[TOPK];
    __shared__ float red[256];
    const int b = blockIdx.x, tid = threadIdx.x;
    if (tid < TOPK) {
        float v = vals[(size_t)b * TOPK + tid];
        sval[tid] = v > 0.0f ? v : 0.0f;
        sind[tid] = inds[(size_t)b * TOPK + tid];
    }
    __syncthreads();

    float acc[3] = {0.f, 0.f, 0.f};
    float xc[3];
#pragma unroll
    for (int j = 0; j < 3; ++j) {
        int d = tid + j * 256;
        xc[j] = x[(size_t)b * D_MODEL + d] - pre_bias[d];
    }
    for (int k = 0; k < TOPK; ++k) {
        float v = sval[k];
        const float* col = decT + (size_t)sind[k] * D_MODEL;
#pragma unroll
        for (int j = 0; j < 3; ++j) acc[j] += v * col[tid + j * 256];
    }
    float err = 0.f;
#pragma unroll
    for (int j = 0; j < 3; ++j) {
        int d = tid + j * 256;
        float r = acc[j] + pre_bias[d];
        recons_out[(size_t)b * D_MODEL + d] = r;
        float e = r - xc[j];
        err += e * e;
        float t = xc[j] - r + pre_bias[d];
        target[(size_t)b * D_MODEL + d] = t;
        atomicAdd(&musum[d], t);
    }
    red[tid] = err;
    __syncthreads();
    for (int s = 128; s > 0; s >>= 1) {
        if (tid < s) red[tid] += red[tid + s];
        __syncthreads();
    }
    if (tid == 0) atomicAdd(&scalars[1], red[0]);
}

// ---------------------------------------------------------------------------
// auxk_recons ; num += (auxk_recons - target)^2 ; den += (mu - target)^2
// ---------------------------------------------------------------------------
__global__ __launch_bounds__(256) void sae_decode_aux(
    const float* __restrict__ pre_bias, const float* __restrict__ decT,
    const float* __restrict__ avals, const int* __restrict__ ainds,
    const float* __restrict__ target, const float* __restrict__ musum,
    float* __restrict__ scalars /* [2]=num, [3]=den */)
{
    __shared__ float sval[AUXK];
    __shared__ int   sind[AUXK];
    __shared__ float redn[256];
    __shared__ float redd[256];
    const int b = blockIdx.x, tid = threadIdx.x;
    if (tid < AUXK) {
        float v = avals[(size_t)b * AUXK + tid];
        sval[tid] = v > 0.0f ? v : 0.0f;
        sind[tid] = ainds[(size_t)b * AUXK + tid];
    }
    __syncthreads();

    float acc[3] = {0.f, 0.f, 0.f};
    for (int k = 0; k < AUXK; ++k) {
        float v = sval[k];
        const float* col = decT + (size_t)sind[k] * D_MODEL;
#pragma unroll
        for (int j = 0; j < 3; ++j) acc[j] += v * col[tid + j * 256];
    }
    float num = 0.f, den = 0.f;
#pragma unroll
    for (int j = 0; j < 3; ++j) {
        int d = tid + j * 256;
        float ar = acc[j] + pre_bias[d];
        float t  = target[(size_t)b * D_MODEL + d];
        float mu = musum[d] * (1.0f / (float)BATCH);
        float e  = ar - t;  num += e * e;
        float m  = mu - t;  den += m * m;
    }
    redn[tid] = num; redd[tid] = den;
    __syncthreads();
    for (int s = 128; s > 0; s >>= 1) {
        if (tid < s) { redn[tid] += redn[tid + s]; redd[tid] += redd[tid + s]; }
        __syncthreads();
    }
    if (tid == 0) {
        atomicAdd(&scalars[2], redn[0]);
        atomicAdd(&scalars[3], redd[0]);
    }
}

// ---------------------------------------------------------------------------
// total_loss = mse + AUXK_COEF * nan_to_num(num/den) ; l0 = l0sum / BATCH
// ---------------------------------------------------------------------------
__global__ void sae_finalize(const float* __restrict__ scalars,
                             float* __restrict__ out_scalars) {
    if (blockIdx.x == 0 && threadIdx.x == 0) {
        float mse  = scalars[1] / (float)((size_t)BATCH * D_MODEL);
        float nmse = scalars[2] / scalars[3];
        if (nmse != nmse) nmse = 0.0f;
        out_scalars[0] = mse + AUXK_COEF * nmse;    // total_loss
        out_scalars[1] = scalars[0] / (float)BATCH; // l0
    }
}

// ---------------------------------------------------------------------------
extern "C" void kernel_launch(void* const* d_in, const int* in_sizes, int n_in,
                              void* d_out, int out_size, void* d_ws, size_t ws_size,
                              hipStream_t stream)
{
    const float* x           = (const float*)d_in[0];
    const float* enc_w       = (const float*)d_in[1];
    const float* dec_w       = (const float*)d_in[2];
    const float* pre_bias    = (const float*)d_in[3];
    const float* latent_bias = (const float*)d_in[4];
    const int*   stats_in    = (const int*)d_in[5];

    float* out         = (float*)d_out;
    float* out_recons  = out;                                   // [BATCH*D_MODEL]
    float* out_scalars = out + (size_t)BATCH * D_MODEL;         // total_loss, l0
    float* out_stats   = out_scalars + 2;                       // [N_DIRS]

    char*  ws  = (char*)d_ws;
    size_t off = 0;
    auto alloc = [&](size_t bytes) -> char* {
        char* p = ws + off;
        off += (bytes + 255) & ~(size_t)255;
        return p;
    };
    float*  latents = (float*) alloc((size_t)BATCH * N_DIRS * 4);   // 256 MB
    float*  decT    = (float*) alloc((size_t)N_DIRS * D_MODEL * 4); // 96 MB
    __bf16* enc_bf  = (__bf16*)alloc((size_t)N_DIRS * D_MODEL * 2); // 48 MB
    __bf16* xc_bf   = (__bf16*)alloc((size_t)BATCH * D_MODEL * 2);  // 3 MB
    float*  vals    = (float*) alloc((size_t)BATCH * TOPK * 4);
    int*    inds    = (int*)   alloc((size_t)BATCH * TOPK * 4);
    float*  avals   = (float*) alloc((size_t)BATCH * AUXK * 4);
    int*    ainds   = (int*)   alloc((size_t)BATCH * AUXK * 4);
    float*  target  = (float*) alloc((size_t)BATCH * D_MODEL * 4);  // 6 MB
    float*  dead    = (float*) alloc((size_t)N_DIRS * 4);
    // contiguous zero-init region: hits | musum | scalars[l0sum,msesum,num,den]
    const int zero_elems = N_DIRS + D_MODEL + 4;
    float* zero_base = (float*)alloc((size_t)zero_elems * 4);
    int*   hits    = (int*)zero_base;
    float* musum   = zero_base + N_DIRS;
    float* scalars = musum + D_MODEL;

    sae_zero<<<(zero_elems + 255) / 256, 256, 0, stream>>>(zero_base, zero_elems);

    const long long enc_elems = (long long)N_DIRS * D_MODEL;
    sae_cvt_bf16<<<(int)((enc_elems / 4 + 255) / 256), 256, 0, stream>>>(
        enc_w, enc_bf, enc_elems);
    sae_make_xc<<<BATCH, 256, 0, stream>>>(x, pre_bias, xc_bf);

    sae_transpose<<<dim3(N_DIRS / 32, D_MODEL / 32), 256, 0, stream>>>(dec_w, decT);

    sae_gemm_latents<<<dim3(N_DIRS / 512, BATCH / 16), 256, 0, stream>>>(
        xc_bf, enc_bf, latent_bias, latents);

    const size_t topk_lds = (size_t)(N_DIRS + 512) * 4;  // row + reduction arrays
    sae_row_topk<<<BATCH, 256, topk_lds, stream>>>(
        latents, nullptr, vals, inds, TOPK, hits, scalars /* l0sum = scalars[0] */);

    sae_stats<<<(N_DIRS + 255) / 256, 256, 0, stream>>>(stats_in, hits, out_stats, dead);

    sae_row_topk<<<BATCH, 256, topk_lds, stream>>>(
        latents, dead, avals, ainds, AUXK, nullptr, nullptr);

    sae_decode_main<<<BATCH, 256, 0, stream>>>(
        x, pre_bias, decT, vals, inds, out_recons, target, musum, scalars);

    sae_decode_aux<<<BATCH, 256, 0, stream>>>(
        pre_bias, decT, avals, ainds, target, musum, scalars);

    sae_finalize<<<1, 1, 0, stream>>>(scalars, out_scalars);
}
